// BinaryAttention_80771154968847
// MI455X (gfx1250) — 16-bit and 8-bit matrix operand order repaired and checked by reading, compiled, not run on hardware
//
#include <hip/hip_runtime.h>

typedef __attribute__((ext_vector_type(16))) _Float16 v16h;
typedef __attribute__((ext_vector_type(8)))  _Float16 v8h;
typedef __attribute__((ext_vector_type(8)))  float    v8f;
typedef __attribute__((ext_vector_type(8)))  int      v8i;

#define NB    16
#define NH    12
#define HD    64
#define CDIM  768
#define C3    2304
#define NSEQ  576
#define BN    (NB * NSEQ)
#define HEADE (NSEQ * HD)
#define NHEAD (NB * NH)
#define TOTE  ((size_t)NHEAD * HEADE)

#define OFF_Q   0ull
#define OFF_K   28311552ull
#define OFF_V   56623104ull
#define OFF_QS  84934656ull
#define OFF_KS  92012544ull
#define OFF_VQ  99090432ull
#define OFF_SQ  106168320ull
#define OFF_SK  (OFF_SQ + 768ull)
#define OFF_SV  (OFF_SK + 768ull)
#define OFF_IV  (OFF_SV + 49152ull)
#define OFF_CTX OFF_Q

#define QKC 64
#define QKP (QKC + 8)
__global__ __launch_bounds__(256) void qkv_gemm(const float* __restrict__ x,
                                                const float* __restrict__ wqkv,
                                                float* __restrict__ q,
                                                float* __restrict__ k,
                                                float* __restrict__ v) {
  __shared__ __align__(16) _Float16 sA[16][QKP];
  const int tid  = threadIdx.x;
  const int lane = tid & 31, wv = tid >> 5, hi = lane >> 4, col = lane & 15;
  const int mtile = blockIdx.x * 16;
  const int b  = mtile / NSEQ;
  const int n0 = mtile - b * NSEQ;
  const int oc0 = blockIdx.y * 256 + wv * 32 + col;
  const int oc1 = oc0 + 16;
  const float* wrow0 = wqkv + (size_t)oc0 * CDIM;
  const float* wrow1 = wqkv + (size_t)oc1 * CDIM;
  const float* xb = x + (size_t)b * CDIM * NSEQ + n0;

  v8f acc0 = {}, acc1 = {};
  for (int st = 0; st < CDIM / QKC; ++st) {
    __syncthreads();
#pragma unroll
    for (int j = 0; j < 4; ++j) {
      int e2 = tid + j * 256;
      int c = e2 >> 4, nn = e2 & 15;
      sA[nn][c] = (_Float16)xb[(size_t)(st * QKC + c) * NSEQ + nn];
    }
    __syncthreads();
#pragma unroll
    for (int k2 = 0; k2 < 2; ++k2) {
      const int kb = k2 * 32 + hi * 16;
      v16h a;
      {
        v8h lo  = *reinterpret_cast<const v8h*>(&sA[col][kb]);
        v8h hi8 = *reinterpret_cast<const v8h*>(&sA[col][kb + 8]);
#pragma unroll
        for (int e = 0; e < 8; ++e) { a[e] = lo[e]; a[e + 8] = hi8[e]; }
      }
      const int cb = st * QKC + k2 * 32 + (hi ? 16 : 0);
      v16h b0, b1;
      const float4* w0 = reinterpret_cast<const float4*>(wrow0 + cb);
      const float4* w1 = reinterpret_cast<const float4*>(wrow1 + cb);
#pragma unroll
      for (int t = 0; t < 4; ++t) {
        float4 f = w0[t], g = w1[t];
        b0[4*t+0]=(_Float16)f.x; b0[4*t+1]=(_Float16)f.y;
        b0[4*t+2]=(_Float16)f.z; b0[4*t+3]=(_Float16)f.w;
        b1[4*t+0]=(_Float16)g.x; b1[4*t+1]=(_Float16)g.y;
        b1[4*t+2]=(_Float16)g.z; b1[4*t+3]=(_Float16)g.w;
      }
      acc0 = __builtin_amdgcn_wmma_f32_16x16x32_f16(false, a, false, b0,
                                                    (short)0, acc0, false, false);
      acc1 = __builtin_amdgcn_wmma_f32_16x16x32_f16(false, a, false, b1,
                                                    (short)0, acc1, false, false);
    }
    if (st + 1 < CDIM / QKC) {
      __builtin_prefetch(wrow0 + (st + 1) * QKC, 0, 1);
      __builtin_prefetch(wrow1 + (st + 1) * QKC, 0, 1);
    }
  }
#pragma unroll
  for (int t2 = 0; t2 < 2; ++t2) {
    const int oc = t2 ? oc1 : oc0;
    const v8f acc = t2 ? acc1 : acc0;
    const int which = oc / CDIM;
    const int hh    = (oc % CDIM) / HD;
    const int d     = oc & (HD - 1);
    float* dst = (which == 0) ? q : (which == 1) ? k : v;
#pragma unroll
    for (int r = 0; r < 8; ++r) {
      int nn = n0 + r + hi * 8;
      dst[(((size_t)(b * NH + hh)) * NSEQ + nn) * HD + d] = acc[r];
    }
  }
}

__global__ __launch_bounds__(256) void head_stats(const float* __restrict__ q,
                                                  const float* __restrict__ k,
                                                  const float* __restrict__ v,
                                                  float* __restrict__ sq,
                                                  float* __restrict__ sk,
                                                  float* __restrict__ sv,
                                                  float* __restrict__ iv) {
  __shared__ float red[256];
  const int hh = blockIdx.x;
  const int tid = threadIdx.x;
  const size_t base = (size_t)hh * HEADE;
  float aq = 0.f, ak = 0.f;
  for (int i = tid; i < HEADE; i += 256) {
    aq += fabsf(q[base + i]);
    ak += fabsf(k[base + i]);
  }
  red[tid] = aq; __syncthreads();
  for (int s = 128; s > 0; s >>= 1) { if (tid < s) red[tid] += red[tid + s]; __syncthreads(); }
  if (tid == 0) sq[hh] = red[0] / (float)HEADE;
  __syncthreads();
  red[tid] = ak; __syncthreads();
  for (int s = 128; s > 0; s >>= 1) { if (tid < s) red[tid] += red[tid + s]; __syncthreads(); }
  if (tid == 0) sk[hh] = red[0] / (float)HEADE;
  __syncthreads();
  const int d = tid & 63, g = tid >> 6;
  float m = 0.f;
  for (int n = g; n < NSEQ; n += 4)
    m = fmaxf(m, fabsf(v[base + (size_t)n * HD + d]));
  red[tid] = m; __syncthreads();
  if (tid < 64) {
    float mm = fmaxf(fmaxf(red[tid], red[tid + 64]), fmaxf(red[tid + 128], red[tid + 192]));
    float s  = 127.f / (mm + 1e-6f);
    sv[hh * HD + tid] = s;
    iv[hh * HD + tid] = 1.f / (s + 1e-6f);
  }
}

__global__ __launch_bounds__(256) void quantize_k(const float* __restrict__ q,
                                                  const float* __restrict__ k,
                                                  const float* __restrict__ v,
                                                  const float* __restrict__ sv,
                                                  char* __restrict__ qs,
                                                  char* __restrict__ ks,
                                                  char* __restrict__ vqT) {
  const size_t i4 = ((size_t)blockIdx.x * 256 + threadIdx.x) * 4;
  if (i4 >= TOTE) return;
  const int hh  = (int)(i4 / HEADE);
  const int rem = (int)(i4 - (size_t)hh * HEADE);
  const int n = rem >> 6, d = rem & 63;
  float4 qv = *reinterpret_cast<const float4*>(q + i4);
  float4 kv = *reinterpret_cast<const float4*>(k + i4);
  float4 vv = *reinterpret_cast<const float4*>(v + i4);
  float4 sc = *reinterpret_cast<const float4*>(sv + hh * HD + d);
  char4 qb, kb;
  qb.x = (qv.x >= 0.f) ? 1 : -1; qb.y = (qv.y >= 0.f) ? 1 : -1;
  qb.z = (qv.z >= 0.f) ? 1 : -1; qb.w = (qv.w >= 0.f) ? 1 : -1;
  kb.x = (kv.x >= 0.f) ? 1 : -1; kb.y = (kv.y >= 0.f) ? 1 : -1;
  kb.z = (kv.z >= 0.f) ? 1 : -1; kb.w = (kv.w >= 0.f) ? 1 : -1;
  *reinterpret_cast<char4*>(qs + i4) = qb;
  *reinterpret_cast<char4*>(ks + i4) = kb;
  char* vt = vqT + (size_t)hh * HEADE + (size_t)d * NSEQ + n;
  vt[0 * NSEQ] = (char)(int)rintf(vv.x * sc.x);
  vt[1 * NSEQ] = (char)(int)rintf(vv.y * sc.y);
  vt[2 * NSEQ] = (char)(int)rintf(vv.z * sc.z);
  vt[3 * NSEQ] = (char)(int)rintf(vv.w * sc.w);
}

#define LDSW 584
__global__ __launch_bounds__(128) void attn_kernel(const char* __restrict__ qs,
                                                   const char* __restrict__ ks,
                                                   const char* __restrict__ vqT,
                                                   const float* __restrict__ sq,
                                                   const float* __restrict__ sk,
                                                   const float* __restrict__ iv,
                                                   float* __restrict__ ctx) {
  __shared__ __align__(16) float         s_attn[16][LDSW];
  __shared__ __align__(16) unsigned char s_pq[16][LDSW];
  __shared__ float s_red[16][8];

  const int hh   = blockIdx.y;
  const int nt   = blockIdx.x;
  const int lane = threadIdx.x & 31;
  const int wv   = threadIdx.x >> 5;
  const int hi   = lane >> 4;
  const int col  = lane & 15;
  const size_t hb = (size_t)hh * HEADE;
  const float alpha = sq[hh] * sk[hh] * 0.125f;

  v8i afrag;
  {
    const char* ap = qs + hb + (size_t)(nt * 16 + col) * HD + (hi ? 8 : 0);
    int2 p0 = *reinterpret_cast<const int2*>(ap +  0);
    int2 p1 = *reinterpret_cast<const int2*>(ap + 16);
    int2 p2 = *reinterpret_cast<const int2*>(ap + 32);
    int2 p3 = *reinterpret_cast<const int2*>(ap + 48);
    afrag[0] = p0.x; afrag[1] = p0.y; afrag[2] = p1.x; afrag[3] = p1.y;
    afrag[4] = p2.x; afrag[5] = p2.y; afrag[6] = p3.x; afrag[7] = p3.y;
  }
  for (int mt = wv; mt < 36; mt += 4) {
    v8i bfrag;
    const char* bp = ks + hb + (size_t)(mt * 16 + col) * HD + (hi ? 8 : 0);
    int2 r0 = *reinterpret_cast<const int2*>(bp);
    int2 r1 = *reinterpret_cast<const int2*>(bp + 16);
    int2 r2 = *reinterpret_cast<const int2*>(bp + 32);
    int2 r3 = *reinterpret_cast<const int2*>(bp + 48);
    bfrag[0] = r0.x; bfrag[1] = r0.y; bfrag[2] = r1.x; bfrag[3] = r1.y;
    bfrag[4] = r2.x; bfrag[5] = r2.y; bfrag[6] = r3.x; bfrag[7] = r3.y;
    v8i c = {};
    c = __builtin_amdgcn_wmma_i32_16x16x64_iu8(true, afrag, true, bfrag,
                                               c, false, false);
#pragma unroll
    for (int r = 0; r < 8; ++r)
      s_attn[r + hi * 8][mt * 16 + col] = alpha * (float)c[r];
  }
  __syncthreads();

  {
    const int row = threadIdx.x >> 3;
    const int sub = threadIdx.x & 7;
    float mx = -3.4e38f;
    for (int j = sub; j < NSEQ; j += 8) mx = fmaxf(mx, s_attn[row][j]);
    s_red[row][sub] = mx; __syncthreads();
    mx = -3.4e38f;
#pragma unroll
    for (int t = 0; t < 8; ++t) mx = fmaxf(mx, s_red[row][t]);
    __syncthreads();
    float sum = 0.f;
    for (int j = sub; j < NSEQ; j += 8) {
      float e = expf(s_attn[row][j] - mx);
      s_attn[row][j] = e;
      sum += e;
    }
    s_red[row][sub] = sum; __syncthreads();
    sum = 0.f;
#pragma unroll
    for (int t = 0; t < 8; ++t) sum += s_red[row][t];
    float inv = 1.f / sum;
    for (int j = sub; j < NSEQ; j += 8) {
      float p = rintf(s_attn[row][j] * inv * 255.f);
      p = fminf(fmaxf(p, 0.f), 255.f);
      s_pq[row][j] = (unsigned char)p;
    }
  }
  __syncthreads();

  {
    const int d = wv * 16 + col;
    const char* bpb = vqT + hb + (size_t)d * NSEQ;
    v8i c = {};
    for (int kt = 0; kt < 9; ++kt) {
      v8i af, bf;
      const unsigned char* pp = &s_pq[col][0] + kt * 64 + (hi ? 8 : 0);
      int2 p0 = *reinterpret_cast<const int2*>(pp +  0);
      int2 p1 = *reinterpret_cast<const int2*>(pp + 16);
      int2 p2 = *reinterpret_cast<const int2*>(pp + 32);
      int2 p3 = *reinterpret_cast<const int2*>(pp + 48);
      af[0] = p0.x; af[1] = p0.y; af[2] = p1.x; af[3] = p1.y;
      af[4] = p2.x; af[5] = p2.y; af[6] = p3.x; af[7] = p3.y;
      const char* bp = bpb + kt * 64 + (hi ? 8 : 0);
      int2 r0 = *reinterpret_cast<const int2*>(bp);
      int2 r1 = *reinterpret_cast<const int2*>(bp + 16);
      int2 r2 = *reinterpret_cast<const int2*>(bp + 32);
      int2 r3 = *reinterpret_cast<const int2*>(bp + 48);
      bf[0] = r0.x; bf[1] = r0.y; bf[2] = r1.x; bf[3] = r1.y;
      bf[4] = r2.x; bf[5] = r2.y; bf[6] = r3.x; bf[7] = r3.y;
      c = __builtin_amdgcn_wmma_i32_16x16x64_iu8(false, af, true, bf,
                                                 c, false, false);
    }
    const int b = hh / NH, h = hh - b * NH;
    const float scd = iv[hh * HD + d] * (1.f / 255.f);
#pragma unroll
    for (int r = 0; r < 8; ++r) {
      int n = nt * 16 + r + hi * 8;
      ctx[((size_t)(b * NSEQ + n)) * CDIM + h * HD + d] = (float)c[r] * scd;
    }
  }
}

__global__ __launch_bounds__(256) void proj_gemm(const float* __restrict__ ctx,
                                                 const float* __restrict__ wproj,
                                                 const float* __restrict__ bias,
                                                 float* __restrict__ out) {
  const int lane  = threadIdx.x & 31;
  const int wv    = threadIdx.x >> 5;
  const int hi    = lane >> 4;
  const int col   = lane & 15;
  const int mtile = blockIdx.x * 16;
  const int oc0   = blockIdx.y * 256 + wv * 32 + col;
  const int oc1   = oc0 + 16;

  const float* arow  = ctx + (size_t)(mtile + col) * CDIM;
  const float* wrow0 = wproj + (size_t)oc0 * CDIM;
  const float* wrow1 = wproj + (size_t)oc1 * CDIM;

  v8f acc0 = {}, acc1 = {};
  for (int kk = 0; kk < CDIM / 32; ++kk) {
    v16h a, b0, b1;
    const float4* ap0 = reinterpret_cast<const float4*>(arow + kk * 32 + (hi ? 16 : 0));
    const float4* ap1 = reinterpret_cast<const float4*>(arow + kk * 32 + (hi ? 16 : 0) + 8);
    float4 f0 = ap0[0], f1 = ap0[1], f2 = ap1[0], f3 = ap1[1];
    a[0]  = (_Float16)f0.x; a[1]  = (_Float16)f0.y; a[2]  = (_Float16)f0.z; a[3]  = (_Float16)f0.w;
    a[4]  = (_Float16)f1.x; a[5]  = (_Float16)f1.y; a[6]  = (_Float16)f1.z; a[7]  = (_Float16)f1.w;
    a[8]  = (_Float16)f2.x; a[9]  = (_Float16)f2.y; a[10] = (_Float16)f2.z; a[11] = (_Float16)f2.w;
    a[12] = (_Float16)f3.x; a[13] = (_Float16)f3.y; a[14] = (_Float16)f3.z; a[15] = (_Float16)f3.w;
    const float4* w0 = reinterpret_cast<const float4*>(wrow0 + kk * 32 + (hi ? 16 : 0));
    const float4* w1 = reinterpret_cast<const float4*>(wrow1 + kk * 32 + (hi ? 16 : 0));
#pragma unroll
    for (int t = 0; t < 4; ++t) {
      float4 f = w0[t], g = w1[t];
      b0[4*t+0]=(_Float16)f.x; b0[4*t+1]=(_Float16)f.y;
      b0[4*t+2]=(_Float16)f.z; b0[4*t+3]=(_Float16)f.w;
      b1[4*t+0]=(_Float16)g.x; b1[4*t+1]=(_Float16)g.y;
      b1[4*t+2]=(_Float16)g.z; b1[4*t+3]=(_Float16)g.w;
    }
    if (kk + 1 < CDIM / 32) {
      __builtin_prefetch(wrow0 + (kk + 1) * 32, 0, 1);
      __builtin_prefetch(wrow1 + (kk + 1) * 32, 0, 1);
    }
    acc0 = __builtin_amdgcn_wmma_f32_16x16x32_f16(false, a, false, b0,
                                                  (short)0, acc0, false, false);
    acc1 = __builtin_amdgcn_wmma_f32_16x16x32_f16(false, a, false, b1,
                                                  (short)0, acc1, false, false);
  }
#pragma unroll
  for (int t2 = 0; t2 < 2; ++t2) {
    const int oc = t2 ? oc1 : oc0;
    const v8f acc = t2 ? acc1 : acc0;
    const float bc = bias[oc];
#pragma unroll
    for (int r = 0; r < 8; ++r) {
      int mr = mtile + r + hi * 8;
      int bb = mr / NSEQ, nn = mr - bb * NSEQ;
      out[((size_t)bb * CDIM + oc) * NSEQ + nn] = acc[r] + bc;
    }
  }
}

extern "C" void kernel_launch(void* const* d_in, const int* in_sizes, int n_in,
                              void* d_out, int out_size, void* d_ws, size_t ws_size,
                              hipStream_t stream) {
  (void)in_sizes; (void)n_in; (void)out_size; (void)ws_size;
  const float* x     = (const float*)d_in[0];
  const float* wqkv  = (const float*)d_in[1];
  const float* wproj = (const float*)d_in[2];
  const float* bproj = (const float*)d_in[3];
  char* ws = (char*)d_ws;
  float* q   = (float*)(ws + OFF_Q);
  float* k   = (float*)(ws + OFF_K);
  float* v   = (float*)(ws + OFF_V);
  char*  qs  = (char*)(ws + OFF_QS);
  char*  ks  = (char*)(ws + OFF_KS);
  char*  vqT = (char*)(ws + OFF_VQ);
  float* sq  = (float*)(ws + OFF_SQ);
  float* sk  = (float*)(ws + OFF_SK);
  float* sv  = (float*)(ws + OFF_SV);
  float* iv  = (float*)(ws + OFF_IV);
  float* ctx = (float*)(ws + OFF_CTX);

  qkv_gemm  <<<dim3(BN / 16, C3 / 256),  256, 0, stream>>>(x, wqkv, q, k, v);
  head_stats<<<dim3(NHEAD),              256, 0, stream>>>(q, k, v, sq, sk, sv, iv);
  quantize_k<<<dim3((int)(TOTE / 1024)), 256, 0, stream>>>(q, k, v, sv, qs, ks, vqT);
  attn_kernel<<<dim3(36, NHEAD),         128, 0, stream>>>(qs, ks, vqT, sq, sk, iv, ctx);
  proj_gemm <<<dim3(BN / 16, CDIM / 256), 256, 0, stream>>>(ctx, wproj, bproj, (float*)d_out);
}
